// BasisSlotAffinityGAT_33268816675310
// MI455X (gfx1250) — matrix operand order repaired and checked by reading, not compiled, not run on hardware
//
#include <hip/hip_runtime.h>
#include <hip/hip_bf16.h>

#define BB 16
#define NN 256
#define DD 768
#define HH 12
#define MM 16
#define KK 16
#define RR 8
#define DH 64
#define TWOD 1536
#define ROWS 4096            /* B*N */
#define QN (BB*HH*NN*NN)     /* 12,582,912 */

typedef __attribute__((ext_vector_type(16))) __bf16 v16bf;
typedef __attribute__((ext_vector_type(8)))  float  v8f;

union FragU { uint4 u[2]; v16bf v; };

__device__ __forceinline__ unsigned short f2bf(float f) {
  unsigned int u = __float_as_uint(f);
  u += 0x7FFFu + ((u >> 16) & 1u);          // round-to-nearest-even
  return (unsigned short)(u >> 16);
}

// ---------------------------------------------------------------- prep
__global__ void k_transpose_bf(const float* __restrict__ W, unsigned short* __restrict__ Wt,
                               int Kdim, int Ncols) {
  size_t idx = (size_t)blockIdx.x * blockDim.x + threadIdx.x;
  if (idx >= (size_t)Kdim * Ncols) return;
  int n = (int)(idx / Kdim), k = (int)(idx % Kdim);
  Wt[idx] = f2bf(W[(size_t)k * Ncols + n]);
}

__global__ void k_build_xcat(const float* __restrict__ desc, const float* __restrict__ nv,
                             unsigned short* __restrict__ xcat) {
  size_t idx = (size_t)blockIdx.x * blockDim.x + threadIdx.x;
  if (idx >= (size_t)ROWS * TWOD) return;
  int row = (int)(idx / TWOD), c = (int)(idx % TWOD);
  float v = (c < DD) ? desc[(size_t)row*DD + c] : nv[(size_t)row*DD + (c - DD)];
  xcat[idx] = f2bf(v);
}

// ---------------------------------------------------------------- WMMA GEMM (double-buffered)
// C[row,col] = act(A[row,:] . Bt[col,:] + bias[col]); A,Bt bf16 row-major (Bt = B^T)
// block = 128 threads = 4 waves in a 2x2 grid; wave tile 32x64 (2 A-frag x 4 B-frag = 8 WMMA)
__device__ __forceinline__ void ldfragA(const unsigned short* p, FragU& f) {
  f.u[0] = *(const uint4*)(p);        // K: +half*16 .. +7
  f.u[1] = *(const uint4*)(p + 8);    // K: .. +15
}
__device__ __forceinline__ void ldfragB(const unsigned short* p, FragU& f) {
  f.u[0] = *(const uint4*)(p);        // K: +half*16 .. +7
  f.u[1] = *(const uint4*)(p + 8);    // K: .. +15
}

__global__ __launch_bounds__(128)
void k_gemm_bf16(const unsigned short* __restrict__ A, const unsigned short* __restrict__ Bt,
                 const float* __restrict__ bias, unsigned short* __restrict__ outBf,
                 float* __restrict__ outF, int Ncols, int Kdim, int relu) {
  const int wave = threadIdx.x >> 5;
  const int lane = threadIdx.x & 31;
  const int half = lane >> 4;
  const int r    = lane & 15;
  const int rowBase = blockIdx.y * 64 + (wave >> 1) * 32;
  const int colBase = blockIdx.x * 128 + (wave & 1) * 64;
  v8f acc[2][4] = {};
  const unsigned short* ap0 = A + (size_t)(rowBase + r) * Kdim + half * 16;
  const unsigned short* ap1 = ap0 + (size_t)16 * Kdim;
  const unsigned short* bp0 = Bt + (size_t)(colBase +  0 + r) * Kdim + half * 16;
  const unsigned short* bp1 = Bt + (size_t)(colBase + 16 + r) * Kdim + half * 16;
  const unsigned short* bp2 = Bt + (size_t)(colBase + 32 + r) * Kdim + half * 16;
  const unsigned short* bp3 = Bt + (size_t)(colBase + 48 + r) * Kdim + half * 16;

  FragU fa0[2], fb0[4], fa1[2], fb1[4];

  auto loadAll = [&](int k0, FragU fa[2], FragU fb[4]) {
    ldfragA(ap0 + k0, fa[0]);
    ldfragA(ap1 + k0, fa[1]);
    ldfragB(bp0 + k0, fb[0]);
    ldfragB(bp1 + k0, fb[1]);
    ldfragB(bp2 + k0, fb[2]);
    ldfragB(bp3 + k0, fb[3]);
  };
  auto mmAll = [&](FragU fa[2], FragU fb[4]) {
#pragma unroll
    for (int rt = 0; rt < 2; ++rt)
#pragma unroll
      for (int nt = 0; nt < 4; ++nt)
        acc[rt][nt] = __builtin_amdgcn_wmma_f32_16x16x32_bf16(
            false, fa[rt].v, false, fb[nt].v, (short)0, acc[rt][nt], false, false);
  };

  const int iters = Kdim >> 5;          // Kdim is a multiple of 64 (768 / 1536)
  loadAll(0, fa0, fb0);
  for (int it = 0; it < iters; it += 2) {
    if (it + 1 < iters) loadAll((it + 1) << 5, fa1, fb1);   // prefetch into buf1
    mmAll(fa0, fb0);                                        // compute buf0
    if (it + 2 < iters) loadAll((it + 2) << 5, fa0, fb0);   // prefetch into buf0
    if (it + 1 < iters) mmAll(fa1, fb1);                    // compute buf1
  }

#pragma unroll
  for (int rt = 0; rt < 2; ++rt) {
#pragma unroll
    for (int nt = 0; nt < 4; ++nt) {
      const int col = colBase + nt*16 + r;
      const float bv = bias[col];
#pragma unroll
      for (int e = 0; e < 8; ++e) {
        const int row = rowBase + rt*16 + e + half*8;
        float v = acc[rt][nt][e] + bv;
        if (relu) v = fmaxf(v, 0.0f);
        size_t o = (size_t)row * Ncols + col;
        if (outF)  outF[o] = v;
        if (outBf) outBf[o] = f2bf(v);
      }
    }
  }
}

// ---------------------------------------------------------------- slot softmax
__global__ __launch_bounds__(256)
void k_slot(const float* __restrict__ X, const float* __restrict__ Ws,
            const float* __restrict__ bs, float* __restrict__ S) {
  const int wid = threadIdx.x >> 5, lane = threadIdx.x & 31;
  const int row = blockIdx.x * 16 + wid * 2 + (lane >> 4);
  const int k = lane & 15;
  const float* xr = X + (size_t)row * DD;
  float s = 0.f;
  for (int d = 0; d < DD; ++d) s = fmaf(xr[d], Ws[d*MM + k], s);
  s += bs[k];                 // TAU_SLOT == 1
  float mx = s;
#pragma unroll
  for (int off = 1; off < 16; off <<= 1) mx = fmaxf(mx, __shfl_xor(mx, off, 32));
  float e = expf(s - mx);
  float sum = e;
#pragma unroll
  for (int off = 1; off < 16; off <<= 1) sum += __shfl_xor(sum, off, 32);
  S[(size_t)row*KK + k] = e / sum;
}

// ---------------------------------------------------------------- basis: G, DG, g_reg
__global__ __launch_bounds__(256)
void k_basis(const float* __restrict__ Up, float* __restrict__ G,
             float* __restrict__ DG, float* __restrict__ regp) {
  __shared__ float Un[256][RR];
  __shared__ float Gs[MM][256];
  __shared__ float nrmM[MM];
  __shared__ float red[8];
  const int t = threadIdx.x;
  {
    float v[RR]; float nr = 0.f;
#pragma unroll
    for (int r = 0; r < RR; ++r) { v[r] = Up[t*RR + r]; nr += v[r]*v[r]; }
    float inv = 1.f / fmaxf(sqrtf(nr), 1e-12f);
#pragma unroll
    for (int r = 0; r < RR; ++r) Un[t][r] = v[r] * inv;
  }
  __syncthreads();
  for (int e = 0; e < 16; ++e) {
    int idx = t*16 + e;
    int m = idx >> 8, k = (idx >> 4) & 15, j = idx & 15;
    float d = 0.f;
#pragma unroll
    for (int r = 0; r < RR; ++r) d += Un[m*16+k][r] * Un[m*16+j][r];
    d = fminf(fmaxf(d, -1.f), 1.f);
    float offd = (k == j) ? 0.f : 1.f;
    float g = fminf(fmaxf((d*offd + 1.f)*0.5f, 0.f), 1.f);
    G[idx] = g;
    Gs[m][k*16 + j] = g;
    DG[idx] = fminf(fmaxf(0.5f*(1.f - d)*offd, 0.f), 1.f);
  }
  __syncthreads();
  if (t < MM) {
    float s = 0.f;
    for (int i = 0; i < 256; ++i) s += Gs[t][i]*Gs[t][i];
    nrmM[t] = fmaxf(sqrtf(s), 1e-8f);
  }
  __syncthreads();
  int a = t >> 4, b = t & 15;
  float dot = 0.f;
  for (int i = 0; i < 256; ++i) dot += Gs[a][i]*Gs[b][i];
  float contrib = (a == b) ? 0.f : dot / (nrmM[a]*nrmM[b]);
#pragma unroll
  for (int off = 16; off >= 1; off >>= 1) contrib += __shfl_xor(contrib, off, 32);
  if ((t & 31) == 0) red[t >> 5] = contrib;
  __syncthreads();
  if (t == 0) {
    float s = 0.f;
    for (int w = 0; w < 8; ++w) s += red[w];
    regp[0] = 0.1f * s / (float)(MM * (MM - 1));
  }
}

// ---------------------------------------------------------------- usage + orth regs
__global__ __launch_bounds__(1024)
void k_usage_orth(const float* __restrict__ S, float* __restrict__ regp) {
  __shared__ float up[MM][64];
  __shared__ float us[MM];
  __shared__ float red[32];
  const int t = threadIdx.x;
  {
    int k = t >> 6, s = t & 63;
    float part = 0.f;
    for (int rw = s; rw < ROWS; rw += 64) part += S[(size_t)rw*KK + k];
    up[k][s] = part;
  }
  __syncthreads();
  if (t < MM) {
    float u = 0.f;
    for (int i = 0; i < 64; ++i) u += up[t][i];
    u /= (float)ROWS;
    us[t] = u * logf(u * (float)MM + 1e-8f);
  }
  __syncthreads();
  if (t == 0) {
    float s = 0.f;
    for (int i = 0; i < MM; ++i) s += us[i];
    regp[1] = 0.01f * s;
  }
  float acc2 = 0.f;
  for (int e = 0; e < 4; ++e) {
    int id = t*4 + e;
    int b = id >> 8, k = (id >> 4) & 15, j = id & 15;
    float acc = 0.f;
    const float* Sb = S + (size_t)b*NN*KK;
    for (int n = 0; n < NN; ++n) acc = fmaf(Sb[n*KK + k], Sb[n*KK + j], acc);
    acc /= (float)NN;
    float d = acc - ((k == j) ? (1.f/(float)MM) : 0.f);
    acc2 += d*d;
  }
#pragma unroll
  for (int off = 16; off >= 1; off >>= 1) acc2 += __shfl_xor(acc2, off, 32);
  if ((t & 31) == 0) red[t >> 5] = acc2;
  __syncthreads();
  if (t == 0) {
    float s = 0.f;
    for (int w = 0; w < 32; ++w) s += red[w];
    regp[2] = 0.01f * s / (float)(BB*MM*MM);
  }
}

// ---------------------------------------------------------------- Q_slot raw
__global__ __launch_bounds__(256)
void k_qslot(const float* __restrict__ S, const float* __restrict__ G,
             float* __restrict__ Qslot) {
  __shared__ float Ssh[NN][17];
  __shared__ float SG[NN][17];
  __shared__ float Gsh[KK][KK];
  const int t = threadIdx.x;
  const int b = blockIdx.x >> 4, m = blockIdx.x & 15;
  for (int idx = t; idx < NN*KK; idx += 256)
    Ssh[idx >> 4][idx & 15] = S[(size_t)(b*NN + (idx >> 4))*KK + (idx & 15)];
  Gsh[t >> 4][t & 15] = G[m*256 + t];
  __syncthreads();
  {
    float sg[KK];
#pragma unroll
    for (int k = 0; k < KK; ++k) sg[k] = 0.f;
    for (int l = 0; l < KK; ++l) {
      float sv = Ssh[t][l];
#pragma unroll
      for (int k = 0; k < KK; ++k) sg[k] = fmaf(sv, Gsh[l][k], sg[k]);
    }
#pragma unroll
    for (int k = 0; k < KK; ++k) SG[t][k] = sg[k];
  }
  __syncthreads();
  float* out = Qslot + (size_t)blockIdx.x * (NN*NN);
  float sr[KK];
#pragma unroll
  for (int k = 0; k < KK; ++k) sr[k] = Ssh[t][k];
  for (int i = 0; i < NN; ++i) {
    float q = 0.f;
#pragma unroll
    for (int k = 0; k < KK; ++k) q = fmaf(SG[i][k], sr[k], q);
    out[i*NN + t] = q;
  }
}

// ---------------------------------------------------------------- affinity-normalize + bistochastic
__global__ __launch_bounds__(1024)
void k_bisto(float* __restrict__ Qslot) {
  __shared__ float rpart[NN][4];
  __shared__ float cpart[4][NN];
  __shared__ float rvec[NN];
  __shared__ float cvec[NN];
  __shared__ float rs[NN];
  float* Q = Qslot + (size_t)blockIdx.x * (NN*NN);
  const int t = threadIdx.x;
  for (int idx = t; idx < NN*NN; idx += 1024) {          // symmetrize + clamp 1e-8
    int i = idx >> 8, j = idx & 255;
    if (i < j) {
      float a = Q[idx], bv = Q[j*NN + i];
      float v = fmaxf(0.5f*(a + bv), 1e-8f);
      Q[idx] = v; Q[j*NN + i] = v;
    } else if (i == j) Q[idx] = fmaxf(Q[idx], 1e-8f);
  }
  __syncthreads();
  {
    int i = t >> 2, qd = t & 3;
    float s = 0.f;
    const float* qr = Q + i*NN + qd*64;
    for (int j = 0; j < 64; ++j) s += qr[j];
    rpart[i][qd] = s;
  }
  __syncthreads();
  if (t < NN) rs[t] = rpart[t][0] + rpart[t][1] + rpart[t][2] + rpart[t][3];
  __syncthreads();
  for (int idx = t; idx < NN*NN; idx += 1024) {          // affinity row-normalize, then clamp 1e-6
    int i = idx >> 8;
    Q[idx] = fmaxf(Q[idx] / fmaxf(rs[i], 1e-8f), 1e-6f);
  }
  if (t < NN) { rvec[t] = 1.f; cvec[t] = 1.f; }
  __syncthreads();
  for (int iter = 0; iter < 10; ++iter) {                // Q0 read-only; scale vectors only
    {
      int i = t >> 2, qd = t & 3;
      float s = 0.f;
      const float* qr = Q + i*NN + qd*64;
      const float* cv = cvec + qd*64;
      for (int j = 0; j < 64; ++j) s = fmaf(qr[j], cv[j], s);
      rpart[i][qd] = s;
    }
    __syncthreads();
    if (t < NN) {
      float raw = rpart[t][0] + rpart[t][1] + rpart[t][2] + rpart[t][3];
      rvec[t] = rvec[t] / (rvec[t]*raw + 1e-6f);
    }
    __syncthreads();
    {
      int j = t & 255, ch = t >> 8;
      float s = 0.f;
      for (int i = ch*64; i < ch*64 + 64; ++i) s = fmaf(Q[i*NN + j], rvec[i], s);
      cpart[ch][j] = s;
    }
    __syncthreads();
    if (t < NN) {
      float raw = cpart[0][t] + cpart[1][t] + cpart[2][t] + cpart[3][t];
      cvec[t] = cvec[t] / (cvec[t]*raw + 1e-6f);
    }
    __syncthreads();
  }
  for (int idx = t; idx < NN*NN; idx += 1024) {
    int i = idx >> 8, j = idx & 255;
    Q[idx] = rvec[i]*Q[idx]*cvec[j];
  }
}

// ---------------------------------------------------------------- per-head DP
__global__ __launch_bounds__(1024)
void k_dp(const float* __restrict__ X, float* __restrict__ DPbuf) {
  __shared__ float xs[NN][DH + 1];
  __shared__ float mxp[NN][4];
  __shared__ float smp[NN][4];
  __shared__ float rowM[NN];
  __shared__ float rowS[NN];
  __shared__ float rs[NN];
  const int t = threadIdx.x;
  const int b = blockIdx.x / HH, h = blockIdx.x % HH;
  for (int idx = t; idx < NN*DH; idx += 1024) {
    int i = idx >> 6, d = idx & 63;
    xs[i][d] = X[(size_t)(b*NN + i)*DD + h*DH + d];
  }
  __syncthreads();
  float* Lg = DPbuf + (size_t)blockIdx.x * (NN*NN);
  const float scale = 1.0f / sqrtf((float)DD);           // SCALE (TAU_AFF == 1)
  {
    int j = t & 255, ch = t >> 8;
    for (int i = ch*64; i < ch*64 + 64; ++i) {
      float dsum = 0.f;
#pragma unroll 16
      for (int d = 0; d < DH; ++d) dsum = fmaf(xs[i][d], xs[j][d], dsum);
      Lg[i*NN + j] = dsum * scale;
    }
  }
  __syncthreads();
  {
    int i = t >> 2, qd = t & 3;                          // online softmax stats
    float mvl = -3.4e38f, ssum = 0.f;
    const float* lr = Lg + i*NN + qd*64;
    for (int j = 0; j < 64; ++j) {
      float v = lr[j];
      if (v > mvl) { ssum = ssum * expf(mvl - v) + 1.f; mvl = v; }
      else ssum += expf(v - mvl);
    }
    mxp[i][qd] = mvl; smp[i][qd] = ssum;
  }
  __syncthreads();
  if (t < NN) {
    float M = fmaxf(fmaxf(mxp[t][0], mxp[t][1]), fmaxf(mxp[t][2], mxp[t][3]));
    float Ss = 0.f;
    for (int q = 0; q < 4; ++q) Ss += smp[t][q]*expf(mxp[t][q] - M);
    rowM[t] = M; rowS[t] = Ss;
  }
  __syncthreads();
  for (int idx = t; idx < NN*NN; idx += 1024) {          // P
    int i = idx >> 8;
    Lg[idx] = expf(Lg[idx] - rowM[i]) / rowS[i];
  }
  __syncthreads();
  for (int idx = t; idx < NN*NN; idx += 1024) {          // symmetrize + clamp 1e-8
    int i = idx >> 8, j = idx & 255;
    if (i < j) {
      float a = Lg[idx], bv = Lg[j*NN + i];
      float v = fmaxf(0.5f*(a + bv), 1e-8f);
      Lg[idx] = v; Lg[j*NN + i] = v;
    } else if (i == j) Lg[idx] = fmaxf(Lg[idx], 1e-8f);
  }
  __syncthreads();
  {
    int i = t >> 2, qd = t & 3;
    float s = 0.f;
    const float* qr = Lg + i*NN + qd*64;
    for (int j = 0; j < 64; ++j) s += qr[j];
    mxp[i][qd] = s;
  }
  __syncthreads();
  if (t < NN) rs[t] = mxp[t][0] + mxp[t][1] + mxp[t][2] + mxp[t][3];
  __syncthreads();
  for (int idx = t; idx < NN*NN; idx += 1024) {
    int i = idx >> 8;
    float v = Lg[idx] / fmaxf(rs[i], 1e-8f);
    Lg[idx] = -logf(fmaxf(v, 1e-6f));
  }
}

// ---------------------------------------------------------------- GW outer loop + Sinkhorn, Pi in LDS
__global__ __launch_bounds__(256)
void k_gw(const float* __restrict__ DPbuf, const float* __restrict__ DG,
          float* __restrict__ gwval) {
  __shared__ float Pi[NN][17];
  __shared__ float Tm[NN][17];
  __shared__ float Cm[NN][17];
  __shared__ float Km[NN][17];
  __shared__ float dg[KK][KK];
  __shared__ float massj[NN];
  __shared__ float ml[KK];
  __shared__ float t2[KK];
  __shared__ float vv[KK];
  __shared__ float wred[8][KK];
  __shared__ float gred[8];
  const int n = threadIdx.x;
  const int wave = n >> 5, lane = n & 31;
  const int bh = blockIdx.x >> 4, m = blockIdx.x & 15;
  dg[n >> 4][n & 15] = DG[m*256 + n];
#pragma unroll
  for (int k = 0; k < KK; ++k) Pi[n][k] = 1.0f/(float)(NN*KK);
  const float* DProw = DPbuf + (size_t)bh*(NN*NN) + n*NN;
  __syncthreads();
  float u = 0.f;
  for (int outer = 0; outer < 3; ++outer) {
    float mj = 0.f;
#pragma unroll
    for (int k = 0; k < KK; ++k) mj += Pi[n][k];
    massj[n] = mj;
#pragma unroll
    for (int k = 0; k < KK; ++k) {                       // mass_l partials
      float v = Pi[n][k];
#pragma unroll
      for (int off = 16; off >= 1; off >>= 1) v += __shfl_xor(v, off, 32);
      if (lane == 0) wred[wave][k] = v;
    }
    float pl[KK];
#pragma unroll
    for (int l = 0; l < KK; ++l) pl[l] = Pi[n][l];
    __syncthreads();
    if (n < KK) {
      float s = 0.f;
      for (int w = 0; w < 8; ++w) s += wred[w][n];
      ml[n] = s;
      float tt = 0.f;
#pragma unroll
      for (int l = 0; l < KK; ++l) { float d = dg[n][l]; tt = fmaf(d*d, ml[l], tt); }
      t2[n] = tt;
    }
#pragma unroll
    for (int k = 0; k < KK; ++k) {                       // T[j][k]
      float s = 0.f;
#pragma unroll
      for (int l = 0; l < KK; ++l) s = fmaf(dg[k][l], pl[l], s);
      Tm[n][k] = s;
    }
    __syncthreads();
    float t1 = 0.f;
    float cr[KK];
#pragma unroll
    for (int k = 0; k < KK; ++k) cr[k] = 0.f;
    for (int j = 0; j < NN; ++j) {                       // term1 + cross, single DP pass
      float d = DProw[j];
      t1 = fmaf(d*d, massj[j], t1);
#pragma unroll
      for (int k = 0; k < KK; ++k) cr[k] = fmaf(d, Tm[j][k], cr[k]);
    }
#pragma unroll
    for (int k = 0; k < KK; ++k) {
      float c = fmaxf(t1 + t2[k] - 2.f*cr[k], 0.f);
      Cm[n][k] = c;
      Km[n][k] = fmaxf(expf(-c * 20.0f), 1e-12f);        // eps = 0.05
    }
    if (n < KK) vv[n] = 1.0f/(float)KK;
    __syncthreads();
    for (int it = 0; it < 10; ++it) {
      float den = 0.f;
#pragma unroll
      for (int k = 0; k < KK; ++k) den = fmaf(Km[n][k], vv[k], den);
      u = (1.0f/(float)NN) / (den + 1e-6f);
#pragma unroll
      for (int k = 0; k < KK; ++k) {
        float v = Km[n][k]*u;
#pragma unroll
        for (int off = 16; off >= 1; off >>= 1) v += __shfl_xor(v, off, 32);
        if (lane == 0) wred[wave][k] = v;
      }
      __syncthreads();
      if (n < KK) {
        float s = 0.f;
        for (int w = 0; w < 8; ++w) s += wred[w][n];
        vv[n] = (1.0f/(float)KK) / (s + 1e-6f);
      }
      __syncthreads();
    }
#pragma unroll
    for (int k = 0; k < KK; ++k) Pi[n][k] = u * Km[n][k] * vv[k];
    __syncthreads();
  }
  float gv = 0.f;
#pragma unroll
  for (int k = 0; k < KK; ++k) gv = fmaf(Cm[n][k], Pi[n][k], gv);
#pragma unroll
  for (int off = 16; off >= 1; off >>= 1) gv += __shfl_xor(gv, off, 32);
  if (lane == 0) gred[wave] = gv;
  __syncthreads();
  if (n == 0) {
    float s = 0.f;
    for (int w = 0; w < 8; ++w) s += gred[w];
    gwval[blockIdx.x] = s;
  }
}

// ---------------------------------------------------------------- alpha + total_reg
__global__ __launch_bounds__(256)
void k_alpha(const float* __restrict__ gwval, float* __restrict__ alpha,
             const float* __restrict__ regp, float* __restrict__ out_total) {
  const int t = threadIdx.x;
  if (t == 0) out_total[0] = regp[0] + regp[1] + regp[2];
  if (t >= BB*HH) return;
  float g[MM]; float mu = 0.f;
#pragma unroll
  for (int i = 0; i < MM; ++i) { g[i] = gwval[t*MM + i]; mu += g[i]; }
  mu /= (float)MM;
  float var = 0.f;
#pragma unroll
  for (int i = 0; i < MM; ++i) { float d = g[i]-mu; var += d*d; }
  var /= (float)(MM - 1);
  float sd = fmaxf(sqrtf(var), 1e-6f);
  float lg[MM]; float mx = -3.4e38f;
#pragma unroll
  for (int i = 0; i < MM; ++i) { lg[i] = -((g[i]-mu)/sd)*10.0f; mx = fmaxf(mx, lg[i]); }
  float s = 0.f;
#pragma unroll
  for (int i = 0; i < MM; ++i) { lg[i] = expf(lg[i]-mx); s += lg[i]; }
#pragma unroll
  for (int i = 0; i < MM; ++i) lg[i] /= s;
  float s2 = 0.f;
#pragma unroll
  for (int i = 0; i < MM; ++i) s2 += lg[i];
  s2 = fmaxf(s2, 1e-8f);
#pragma unroll
  for (int i = 0; i < MM; ++i) alpha[t*MM + i] = lg[i]/s2;
}

// ---------------------------------------------------------------- final mixture + log
__global__ __launch_bounds__(256)
void k_combine(const float* __restrict__ Qslot, const float* __restrict__ alpha,
               float* __restrict__ outBias, float* __restrict__ outQ) {
  __shared__ float al[HH][MM];
  const int t = threadIdx.x;
  const int b = blockIdx.x >> 8, i = blockIdx.x & 255;
  if (t < HH*MM) al[t / MM][t % MM] = alpha[(size_t)b*HH*MM + t];
  __syncthreads();
  float v[MM];
#pragma unroll
  for (int m = 0; m < MM; ++m)
    v[m] = Qslot[(((size_t)(b*MM + m)*NN + i)*NN) + t];
#pragma unroll
  for (int h = 0; h < HH; ++h) {
    float q = 0.f;
#pragma unroll
    for (int m = 0; m < MM; ++m) q = fmaf(al[h][m], v[m], q);
    size_t o = (((size_t)(b*HH + h)*NN + i)*NN) + t;
    outQ[o] = q;
    outBias[o] = logf(fmaxf(q, 1e-8f));
  }
}

// ---------------------------------------------------------------- launch
extern "C" void kernel_launch(void* const* d_in, const int* in_sizes, int n_in,
                              void* d_out, int out_size, void* d_ws, size_t ws_size,
                              hipStream_t stream) {
  (void)in_sizes; (void)n_in; (void)out_size; (void)ws_size;
  const float* desc = (const float*)d_in[0];
  const float* nv   = (const float*)d_in[1];
  const float* W1   = (const float*)d_in[2];
  const float* b1   = (const float*)d_in[3];
  const float* W2   = (const float*)d_in[4];
  const float* b2   = (const float*)d_in[5];
  const float* Ws   = (const float*)d_in[6];
  const float* bs   = (const float*)d_in[7];
  const float* Up   = (const float*)d_in[8];
  char* ws = (char*)d_ws;
  // QSLOT region (67.1MB) transiently hosts the bf16 staging buffers
  float*          QSLOT = (float*)(ws + 0);
  unsigned short* XCAT  = (unsigned short*)(ws + 0);
  unsigned short* W1T   = (unsigned short*)(ws + 12582912);
  unsigned short* W2T   = (unsigned short*)(ws + 14942208);
  unsigned short* HBF   = (unsigned short*)(ws + 16121856);
  float* DPb  = (float*)(ws + 67108864);
  float* X    = (float*)(ws + 117440512);
  float* Sb   = (float*)(ws + 130023424);
  float* G    = (float*)(ws + 130285568);
  float* DG   = (float*)(ws + 130301952);
  float* GW   = (float*)(ws + 130318336);
  float* AL   = (float*)(ws + 130330624);
  float* REGP = (float*)(ws + 130342912);
  float* outB = (float*)d_out;
  float* outQ = outB + (size_t)QN;
  float* outT = outB + 2*(size_t)QN;

  k_transpose_bf<<<(768*1536 + 255)/256, 256, 0, stream>>>(W1, W1T, 1536, 768);
  k_transpose_bf<<<(768*768  + 255)/256, 256, 0, stream>>>(W2, W2T, 768, 768);
  k_build_xcat<<<(ROWS*TWOD + 255)/256, 256, 0, stream>>>(desc, nv, XCAT);
  k_gemm_bf16<<<dim3(6, 64), 128, 0, stream>>>(XCAT, W1T, b1, HBF, nullptr, 768, 1536, 1);
  k_gemm_bf16<<<dim3(6, 64), 128, 0, stream>>>(HBF,  W2T, b2, nullptr, X,   768, 768, 0);
  k_slot<<<256, 256, 0, stream>>>(X, Ws, bs, Sb);
  k_basis<<<1, 256, 0, stream>>>(Up, G, DG, REGP);
  k_usage_orth<<<1, 1024, 0, stream>>>(Sb, REGP);
  k_qslot<<<256, 256, 0, stream>>>(Sb, G, QSLOT);
  k_bisto<<<256, 1024, 0, stream>>>(QSLOT);
  k_dp<<<192, 1024, 0, stream>>>(X, DPb);
  k_gw<<<3072, 256, 0, stream>>>(DPb, DG, GW);
  k_alpha<<<1, 256, 0, stream>>>(GW, AL, REGP, outT);
  k_combine<<<4096, 256, 0, stream>>>(QSLOT, AL, outB, outQ);
}
